// VectorQuantizer_50337016709434
// MI455X (gfx1250) — matrix operand order repaired and checked by reading, not compiled, not run on hardware
//
#include <hip/hip_runtime.h>

#define EMB 256
#define K_CODES 512
#define ROWS 64        // spatial rows per workgroup (4 waves x 16 rows)
#define PITCH 132      // dwords per LDS row (264 bf16 slots; bank-conflict pad, 16B aligned)

typedef __attribute__((ext_vector_type(16))) __bf16 v16bf;
typedef __attribute__((ext_vector_type(8)))  float  v8f;

union Frag { uint4 q[2]; v16bf v; };

__device__ __forceinline__ unsigned short f2bf(float f) {
    unsigned u = __float_as_uint(f);
    unsigned r = (u + 0x7FFFu + ((u >> 16) & 1u)) >> 16;  // round-to-nearest-even
    return (unsigned short)r;
}
__device__ __forceinline__ float bf2f(unsigned short h) {
    return __uint_as_float(((unsigned)h) << 16);
}
// Generic pointers to LDS carry the LDS byte offset in their low 32 bits.
__device__ __forceinline__ unsigned lds_off(const void* p) {
    return (unsigned)(uintptr_t)p;
}
// Async global->LDS 16B copy (GVS mode), tracked by ASYNCcnt.
__device__ __forceinline__ void async_cp16(unsigned ldsByteOff, unsigned gByteOff,
                                           const unsigned* base) {
    asm volatile("global_load_async_to_lds_b128 %0, %1, %2 offset:0"
                 :: "v"(ldsByteOff), "v"(gByteOff), "s"(base) : "memory");
}

// ---------------------------------------------------------------------------
// Prep 1: pack codebook into WMMA B-fragment layout (hi/lo bf16 split).
// For (tile, chunk): lane l supplies column n = l%16 (code tile*16+n),
// K rows chunk*32 + (l/16)*16 + [0..15], 2 bf16 per dword, 32B per lane.
// ---------------------------------------------------------------------------
__global__ __launch_bounds__(256) void vq_prep_frags(const float* __restrict__ cb,
                                                     unsigned* __restrict__ hiF,
                                                     unsigned* __restrict__ loF) {
    int g     = blockIdx.x * 256 + threadIdx.x;   // 0..8191 = tile*256 + chunk*32 + lane
    int lane  = g & 31;
    int chunk = (g >> 5) & 7;
    int tile  = g >> 8;
    int code  = tile * 16 + (lane & 15);
    int kbase = chunk * 32 + ((lane >> 4) << 4);
    const float* src = cb + code * EMB + kbase;
    unsigned base = (unsigned)g * 8u;
    #pragma unroll
    for (int i = 0; i < 8; ++i) {
        float a = src[2 * i], b = src[2 * i + 1];
        unsigned short ah = f2bf(a), bh = f2bf(b);
        float al = a - bf2f(ah), bl = b - bf2f(bh);
        hiF[base + i] = (unsigned)ah       | ((unsigned)bh       << 16);
        loF[base + i] = (unsigned)f2bf(al) | ((unsigned)f2bf(bl) << 16);
    }
}

// Prep 2: ||e_k||^2 per code.
__global__ __launch_bounds__(256) void vq_prep_sq(const float* __restrict__ cb,
                                                  float* __restrict__ sq) {
    int code = blockIdx.x * 256 + threadIdx.x;
    if (code < K_CODES) {
        const float* r = cb + code * EMB;
        float s = 0.f;
        #pragma unroll 8
        for (int c = 0; c < EMB; ++c) { float v = r[c]; s = __builtin_fmaf(v, v, s); }
        sq[code] = s;
    }
}

// ---------------------------------------------------------------------------
// Main: 1024 WGs x 128 threads; 64 spatial rows per WG (16 per wave).
// bf16x3 WMMA distance GEMM with async-LDS double-buffered codebook tiles.
// ---------------------------------------------------------------------------
__global__ __launch_bounds__(128) void vq_main(const float* __restrict__ z,
                                               const float* __restrict__ cb,
                                               const unsigned* __restrict__ hiF,
                                               const unsigned* __restrict__ loF,
                                               const float* __restrict__ cbSq,
                                               float* __restrict__ zq,
                                               float* __restrict__ idxOut,
                                               float* __restrict__ partial) {
    __shared__ __align__(16) unsigned sHi[ROWS * PITCH];       // 33792 B
    __shared__ __align__(16) unsigned sLo[ROWS * PITCH];       // 33792 B
    __shared__ __align__(16) unsigned sBufHi[2][2048];         // 16384 B (8 KB per tile)
    __shared__ __align__(16) unsigned sBufLo[2][2048];         // 16384 B
    __shared__ int   sIdx[ROWS];
    __shared__ float sRed[4];

    const int tid = threadIdx.x;
    const int wg  = blockIdx.x;
    // element (r, c) of this tile lives at zBase + c*4096 + r
    const size_t zBase = (size_t)(wg >> 6) * (256u * 4096u) + (size_t)(wg & 63) * 64u;

    // ---- load z tile to LDS as packed bf16 hi/lo pairs (coalesced along r) ----
    {
        const int r  = tid & 63;
        const int ph = tid >> 6;
        for (int it = 0; it < 64; ++it) {
            int cp  = it * 2 + ph;                       // channel pair 0..127
            float a = z[zBase + (size_t)(2 * cp)     * 4096u + r];
            float b = z[zBase + (size_t)(2 * cp + 1) * 4096u + r];
            unsigned short ah = f2bf(a), bh = f2bf(b);
            float al = a - bf2f(ah), bl = b - bf2f(bh);
            sHi[r * PITCH + cp] = (unsigned)ah       | ((unsigned)bh       << 16);
            sLo[r * PITCH + cp] = (unsigned)f2bf(al) | ((unsigned)f2bf(bl) << 16);
        }
    }
    __syncthreads();

    const int lane = tid & 31;
    const int wave = tid >> 5;
    const int half = lane >> 4;
    const int ln   = lane & 15;
    const int rowA = wave * 16 + ln;     // A row this lane feeds

    // ---- A fragments -> registers, once (barriers below prevent hoisting) ----
    Frag aH[8], aL[8];
    #pragma unroll
    for (int chunk = 0; chunk < 8; ++chunk) {
        int ab = rowA * PITCH + chunk * 16 + half * 8;
        aH[chunk].q[0] = *(const uint4*)&sHi[ab];
        aH[chunk].q[1] = *(const uint4*)&sHi[ab + 4];
        aL[chunk].q[0] = *(const uint4*)&sLo[ab];
        aL[chunk].q[1] = *(const uint4*)&sLo[ab + 4];
    }

    // ---- async-prefetch codebook tile 0 into buffer 0 (8 ops/wave) ----
    const unsigned myCopyOff = (unsigned)tid * 64u;            // 64 B per thread per array
    const unsigned bufHiOff0 = lds_off(&sBufHi[0][0]);
    const unsigned bufLoOff0 = lds_off(&sBufLo[0][0]);
    {
        unsigned g0 = myCopyOff;                               // tile 0: bytes [tid*64, +64)
        #pragma unroll
        for (int k = 0; k < 4; ++k) {
            async_cp16(bufHiOff0 + myCopyOff + 16u * k, g0 + 16u * k, hiF);
            async_cp16(bufLoOff0 + myCopyOff + 16u * k, g0 + 16u * k, loF);
        }
    }

    float best[8]; int bidx[8];
    #pragma unroll
    for (int j = 0; j < 8; ++j) { best[j] = 3.4e38f; bidx[j] = 0; }

    for (int tile = 0; tile < 32; ++tile) {
        const int cur = tile & 1;
        if (tile + 1 < 32) {           // prefetch next tile into other buffer
            const int nxt = cur ^ 1;
            unsigned g0 = (unsigned)(tile + 1) * 8192u + myCopyOff;
            unsigned lh = (nxt ? lds_off(&sBufHi[1][0]) : bufHiOff0) + myCopyOff;
            unsigned ll = (nxt ? lds_off(&sBufLo[1][0]) : bufLoOff0) + myCopyOff;
            #pragma unroll
            for (int k = 0; k < 4; ++k) {
                async_cp16(lh + 16u * k, g0 + 16u * k, hiF);
                async_cp16(ll + 16u * k, g0 + 16u * k, loF);
            }
            asm volatile("s_wait_asynccnt 0x8" ::: "memory");  // current tile's 8 ops done
        } else {
            asm volatile("s_wait_asynccnt 0x0" ::: "memory");
        }
        __syncthreads();               // current buffer visible to all waves

        v8f acc = {0.f, 0.f, 0.f, 0.f, 0.f, 0.f, 0.f, 0.f};
        #pragma unroll
        for (int chunk = 0; chunk < 8; ++chunk) {
            Frag bH, bL;
            const unsigned* ph = &sBufHi[cur][chunk * 256 + lane * 8];
            const unsigned* pl = &sBufLo[cur][chunk * 256 + lane * 8];
            bH.q[0] = *(const uint4*)ph;       bH.q[1] = *(const uint4*)(ph + 4);
            bL.q[0] = *(const uint4*)pl;       bL.q[1] = *(const uint4*)(pl + 4);
            // bf16x3: dot = hi*hi + hi*lo + lo*hi  (lo*lo ~2^-16, negligible)
            acc = __builtin_amdgcn_wmma_f32_16x16x32_bf16(false, aH[chunk].v, false, bH.v, (short)0, acc, false, false);
            acc = __builtin_amdgcn_wmma_f32_16x16x32_bf16(false, aH[chunk].v, false, bL.v, (short)0, acc, false, false);
            acc = __builtin_amdgcn_wmma_f32_16x16x32_bf16(false, aL[chunk].v, false, bH.v, (short)0, acc, false, false);
        }
        int   code = tile * 16 + ln;        // C/D layout: column n = lane%16
        float sq   = cbSq[code];
        #pragma unroll
        for (int j = 0; j < 8; ++j) {       // row m = j + 8*half
            float d = __builtin_fmaf(-2.0f, acc[j], sq);
            if (d < best[j]) { best[j] = d; bidx[j] = code; }
        }
        __syncthreads();               // reads done before this buffer is overwritten
    }

    // argmin across the 16 columns in each half; tie -> smaller index (jnp.argmin)
    #pragma unroll
    for (int j = 0; j < 8; ++j) {
        float v = best[j]; int ix = bidx[j];
        #pragma unroll
        for (int m = 8; m >= 1; m >>= 1) {
            float ov = __shfl_xor(v, m, 16);
            int   oi = __shfl_xor(ix, m, 16);
            if (ov < v || (ov == v && oi < ix)) { v = ov; ix = oi; }
        }
        best[j] = v; bidx[j] = ix;
    }
    if (ln == 0) {
        #pragma unroll
        for (int j = 0; j < 8; ++j) {
            int rl   = wave * 16 + half * 8 + j;    // local row
            sIdx[rl] = bidx[j];
            idxOut[(size_t)wg * ROWS + rl] = (float)bidx[j];
        }
    }
    __syncthreads();

    // ---- scatter codebook[idx] back to [B,C,D,H,W] + loss partial ----
    float lsum = 0.f;
    {
        const int r  = tid & 63;
        const int ph = tid >> 6;
        const float* crow = cb + sIdx[r] * EMB;
        for (int it = 0; it < 128; ++it) {
            int c = it * 2 + ph;
            float cv = crow[c];
            size_t g = zBase + (size_t)c * 4096u + r;
            float d = cv - z[g];
            zq[g] = cv;
            lsum = __builtin_fmaf(d, d, lsum);
        }
    }
    #pragma unroll
    for (int m = 16; m >= 1; m >>= 1) lsum += __shfl_xor(lsum, m, 32);
    if (lane == 0) sRed[wave] = lsum;
    __syncthreads();
    if (tid == 0) partial[wg] = (sRed[0] + sRed[1]) + (sRed[2] + sRed[3]);
}

// Deterministic fixed-order loss reduction (no float atomics).
__global__ __launch_bounds__(256) void vq_loss_reduce(const float* __restrict__ partial,
                                                      float* __restrict__ lossOut) {
    __shared__ float s[256];
    int t = threadIdx.x;
    float a = 0.f;
    #pragma unroll
    for (int k = 0; k < 4; ++k) a += partial[t + 256 * k];   // 1024 partials
    s[t] = a; __syncthreads();
    for (int st = 128; st > 0; st >>= 1) { if (t < st) s[t] += s[t + st]; __syncthreads(); }
    // vq_loss = (1 + 0.25) * mean(diff^2) over 65536*256 elements
    if (t == 0) lossOut[0] = s[0] * (1.25f / 16777216.f);
}

extern "C" void kernel_launch(void* const* d_in, const int* in_sizes, int n_in,
                              void* d_out, int out_size, void* d_ws, size_t ws_size,
                              hipStream_t stream) {
    const float* z  = (const float*)d_in[0];   // [16,256,16,16,16]
    const float* cb = (const float*)d_in[1];   // [512,256]

    float* out    = (float*)d_out;
    float* zq     = out;                       // 16777216 floats
    float* loss   = out + 16777216;            // 1 float
    float* idxOut = out + 16777217;            // 65536 indices (as float)

    unsigned* hiF    = (unsigned*)d_ws;        // 65536 dwords
    unsigned* loF    = hiF + 65536;            // 65536 dwords
    float*    cbSq   = (float*)(loF + 65536);  // 512 floats
    float*    part   = cbSq + 512;             // 1024 floats

    vq_prep_frags<<<32, 256, 0, stream>>>(cb, hiF, loF);
    vq_prep_sq<<<2, 256, 0, stream>>>(cb, cbSq);
    vq_main<<<1024, 128, 0, stream>>>(z, cb, hiF, loF, cbSq, zq, idxOut, part);
    vq_loss_reduce<<<1, 256, 0, stream>>>(part, loss);
}
